// Model_14216341750290
// MI455X (gfx1250) — matrix operand order repaired and checked by reading, not compiled, not run on hardware
//
#include <hip/hip_runtime.h>
#include <hip/hip_bf16.h>

typedef __attribute__((ext_vector_type(16))) _Float16 v16h;
typedef __attribute__((ext_vector_type(8)))  _Float16 v8h;
typedef __attribute__((ext_vector_type(8)))  float    v8f;

#define NBATCH 2048
#define TDIM   15
#define TPAD   24      // time padded by +4 front; covers t+dt in [0,23]
#define VDIM   64
#define HCH    8

// 1/sqrt(1 + 1e-5)
#define BN_RS 0.99999500003749937f

__device__ __forceinline__ v16h cat16(v8h lo, v8h hi) {
  return __builtin_shufflevector(lo, hi, 0,1,2,3,4,5,6,7,8,9,10,11,12,13,14,15);
}

__global__ __launch_bounds__(256, 1) void stgcn_fused(
    const float* __restrict__ x,      const float* __restrict__ Aadj,
    const float* __restrict__ bn0_g,  const float* __restrict__ bn0_b,
    const float* __restrict__ res_w,  const float* __restrict__ res_b,
    const float* __restrict__ res_bn_g, const float* __restrict__ res_bn_b,
    const float* __restrict__ gcn_w,  const float* __restrict__ gcn_b,
    const float* __restrict__ gru_wih, const float* __restrict__ gru_bih,
    const float* __restrict__ gru_bhh,
    const float* __restrict__ bn1_g,  const float* __restrict__ bn1_b,
    const float* __restrict__ tcn_w,  const float* __restrict__ tcn_b,
    const float* __restrict__ bn2_g,  const float* __restrict__ bn2_b,
    const float* __restrict__ fcn_w,  const float* __restrict__ fcn_b,
    const float* __restrict__ clf_w,  const float* __restrict__ clf_b,
    float* __restrict__ out)
{
  // ---- LDS working set (~150 KB; 2 workgroups / WGP on 320 KB) ----
  __shared__ alignas(16) _Float16 s_x   [3][TPAD][VDIM];  // bn0'd input, T-padded
  __shared__ alignas(16) _Float16 s_At  [VDIM][VDIM];     // adjacency TRANSPOSED: s_At[w][v]=A[v][w]
  __shared__ alignas(16) _Float16 s_g0  [128][VDIM];      // 1x1-conv result, rows (o*15+t) pad 128
  __shared__ alignas(16) float    s_gflat[128][VDIM];     // graph-conv result, SAME flat row layout
  __shared__ alignas(16) _Float16 s_y2  [TPAD][VDIM][16]; // tcn input, CHANNEL-INNERMOST
  __shared__ alignas(16) float    s_res [HCH][TDIM][VDIM];// residual; later final y in place
  __shared__ alignas(16) _Float16 s_twT [16][160];        // tcn weights TRANSPOSED: [o][k'=dt*16+ci]
  __shared__ alignas(16) float    s_p   [HCH][VDIM];      // T-pooled features

  const int n    = blockIdx.x;
  const int tid  = threadIdx.x;
  const int lane = tid & 31;
  const int wv   = tid >> 5;
  const int halfsel = lane >> 4;        // which 16-lane half of the wave

  // ================= Stage A: stage inputs / weights into LDS ==============
  for (int i = tid; i < VDIM * VDIM; i += 256) {
    int v = i >> 6, w = i & 63;
    s_At[w][v] = (_Float16)Aadj[i];                  // transpose on the fly
  }

  for (int i = tid; i < 3 * TPAD * VDIM; i += 256) {
    int c = i / (TPAD * VDIM);
    int rem = i % (TPAD * VDIM);
    int tp = rem >> 6, v = rem & 63;
    int t = tp - 4;
    float val = 0.0f;
    if (t >= 0 && t < TDIM) {
      float s = bn0_g[v * 3 + c] * BN_RS;            // channel index = v*3+c after transpose
      val = x[(((size_t)n * 4 + c) * TDIM + t) * VDIM + v] * s + bn0_b[v * 3 + c];
    }
    s_x[c][tp][v] = (_Float16)val;
  }

  for (int i = tid; i < 16 * 160; i += 256) {        // tcn weights, K reordered k' = dt*16+ci
    int o = i / 160, k = i % 160;
    int dt = k >> 4, ci = k & 15;
    float w = 0.0f;
    if (dt < 9 && o < 8) w = tcn_w[(o * 16 + ci) * 9 + dt];
    s_twT[o][k] = (_Float16)w;
  }
  __syncthreads();

  // ================= Stage B: gcn 1x1 conv -> s_g0 (f16, 128x64) ===========
  for (int i = tid; i < 128 * VDIM; i += 256) {
    int r = i >> 6, v = i & 63;
    float val = 0.0f;
    if (r < HCH * TDIM) {
      int o = r / TDIM, t = r % TDIM;
      val = gcn_b[o];
#pragma unroll
      for (int c = 0; c < 3; ++c) val += gcn_w[o * 3 + c] * (float)s_x[c][t + 4][v];
    }
    s_g0[r][v] = (_Float16)val;
  }
  __syncthreads();

  // ===== Stage C: graph conv  g = g0 @ A   (M=128, N=64, K=64) via WMMA ====
  // wave wv owns M-rows [16*wv, 16*wv+16); 4 N-tiles x 2 K-steps = 8 WMMAs/wave.
  // Operands: paired 16B ds_load_b128 (contiguous K-runs). D-store: flat,
  // unguarded, division-free (pad rows land in s_gflat[120..127] slack).
  {
    const int arow = wv * 16 + (lane & 15);
    const int col0 = lane & 15;
    v8f acc[4] = {};
#pragma unroll
    for (int ks = 0; ks < 2; ++ks) {
      const int r0 = ks * 32 + halfsel * 8;          // A K-run start (ISA A 16x32 layout)
      v16h afrag = cat16(*(const v8h*)&s_g0[arow][r0],
                         *(const v8h*)&s_g0[arow][r0 + 16]);
      const int kb = ks * 32 + halfsel * 8;          // B K-run start (ISA layout, matches A)
#pragma unroll
      for (int nt = 0; nt < 4; ++nt) {
        int col = nt * 16 + col0;
        v16h bfrag = cat16(*(const v8h*)&s_At[col][kb],
                           *(const v8h*)&s_At[col][kb + 16]);
        acc[nt] = __builtin_amdgcn_wmma_f32_16x16x32_f16(
            false, afrag, false, bfrag, (short)0, acc[nt], false, false);
      }
    }
    const int rbase = wv * 16 + halfsel * 8;         // C/D f32: M = half*8+i2, N = lane&15
#pragma unroll
    for (int nt = 0; nt < 4; ++nt)
#pragma unroll
      for (int i2 = 0; i2 < 8; ++i2)
        s_gflat[rbase + i2][nt * 16 + col0] = acc[nt][i2];
  }
  __syncthreads();

  // ================= Stage D: residual branch (K=27 conv + BN, VALU) =======
  for (int i = tid; i < HCH * TDIM * VDIM; i += 256) {
    int o = i / (TDIM * VDIM);
    int rem = i % (TDIM * VDIM);
    int t = rem >> 6, v = rem & 63;
    float acc = res_b[o];
#pragma unroll
    for (int c = 0; c < 3; ++c)
#pragma unroll
      for (int dt = 0; dt < 9; ++dt)
        acc += res_w[(o * 3 + c) * 9 + dt] * (float)s_x[c][t + dt][v];
    s_res[o][t][v] = acc * (res_bn_g[o] * BN_RS) + res_bn_b[o];
  }

  // ====== Stage E: GRU gates (h0=0 -> gh=bhh) + assemble tcn input y =======
  for (int i = tid; i < 9 * VDIM * 16; i += 256) {   // zero T-pad borders of s_y2
    int tpi = i / (VDIM * 16);
    int rem = i % (VDIM * 16);
    int tp = (tpi < 4) ? tpi : tpi + 15;             // tp in {0..3, 19..23}
    s_y2[tp][rem >> 4][rem & 15] = (_Float16)0.0f;
  }
  for (int i = tid; i < VDIM * TDIM; i += 256) {
    int v = i / TDIM, t = i % TDIM;
    float gv[8], gi[24];
#pragma unroll
    for (int h = 0; h < 8; ++h) gv[h] = s_gflat[h * TDIM + t][v];  // h*15 is a literal
#pragma unroll
    for (int j = 0; j < 24; ++j) {
      float a = gru_bih[j];
#pragma unroll
      for (int h = 0; h < 8; ++h) a += gv[h] * gru_wih[j * 8 + h];
      gi[j] = a;
    }
#pragma unroll
    for (int h = 0; h < 8; ++h) {                    // channels 0..7 = relu(BN(g))
      float yv = gv[h] * (bn1_g[h] * BN_RS) + bn1_b[h];
      s_y2[t + 4][v][h] = (_Float16)(yv > 0.0f ? yv : 0.0f);
    }
#pragma unroll
    for (int j = 0; j < 8; ++j) {                    // channels 8..15 = relu(BN(hout))
      float r  = 1.0f / (1.0f + __expf(-(gi[j]      + gru_bhh[j])));
      float z  = 1.0f / (1.0f + __expf(-(gi[8 + j]  + gru_bhh[8 + j])));
      float nn = tanhf(gi[16 + j] + r * gru_bhh[16 + j]);
      float hout = (1.0f - z) * nn;                  // + z*h0, h0==0
      float yv = hout * (bn1_g[8 + j] * BN_RS) + bn1_b[8 + j];
      s_y2[t + 4][v][8 + j] = (_Float16)(yv > 0.0f ? yv : 0.0f);
    }
  }
  __syncthreads();

  // ===== Stage F: tcn conv via WMMA (M=16 v's, N=8 ch, K'=dt*16+ci, 160) ===
  // 60 (t, vblock) tile jobs round-robin over 8 waves; 5 K-steps each.
  // K-run starts r0 = ks*32 + halfsel*8 are multiples of 8 -> dt fixed per run,
  // ci-run contiguous in s_y2 => pure ds_load_b128 fragment builds, no branches.
  {
    const int o = lane & 15;
    for (int job = wv; job < 60; job += 8) {
      int t  = job >> 2;
      int vb = job & 3;
      int vrow = vb * 16 + (lane & 15);
      v8f acc = {};
#pragma unroll
      for (int ks = 0; ks < 5; ++ks) {
        const int r0  = ks * 32 + halfsel * 8;       // K'-run start
        const int dt0 = r0 >> 4;                     // tap of first run
        const int ci0 = r0 & 15;                     // channel offset (0 or 8)
        v16h afrag = cat16(*(const v8h*)&s_y2[t + dt0    ][vrow][ci0],
                           *(const v8h*)&s_y2[t + dt0 + 1][vrow][ci0]);
        const int kb = ks * 32 + halfsel * 8;
        v16h bfrag = cat16(*(const v8h*)&s_twT[o][kb],
                           *(const v8h*)&s_twT[o][kb + 16]);
        acc = __builtin_amdgcn_wmma_f32_16x16x32_f16(
            false, afrag, false, bfrag, (short)0, acc, false, false);
      }
      // epilogue: +bias, BN2, +res, leaky-ReLU; overwrite s_res in place
      if (o < 8) {
        float sg = bn2_g[o] * BN_RS, sb = bn2_b[o], tb = tcn_b[o];
#pragma unroll
        for (int i2 = 0; i2 < 8; ++i2) {
          int v = vb * 16 + halfsel * 8 + i2;
          float val = (acc[i2] + tb) * sg + sb + s_res[o][t][v];
          val = val > 0.0f ? val : 0.01f * val;
          s_res[o][t][v] = val;                      // unique (o,t,v) per lane: race-free
        }
      }
    }
  }
  __syncthreads();

  // ================= Stage G: mean over T, then the two heads ==============
  for (int i = tid; i < HCH * VDIM; i += 256) {
    int o = i >> 6, v = i & 63;
    float a = 0.0f;
#pragma unroll
    for (int t = 0; t < TDIM; ++t) a += s_res[o][t][v];
    s_p[o][v] = a * (1.0f / 15.0f);
  }
  __syncthreads();

  if (tid < VDIM) {
    int v = tid;
    float pv[8];
#pragma unroll
    for (int o = 0; o < 8; ++o) pv[o] = s_p[o][v];
    float reg = fcn_b[0];
#pragma unroll
    for (int o = 0; o < 8; ++o) reg += pv[o] * fcn_w[o];
    out[(size_t)n * VDIM + v] = reg;
#pragma unroll
    for (int k2 = 0; k2 < 2; ++k2) {
      float cv = clf_b[k2];
#pragma unroll
      for (int o = 0; o < 8; ++o) cv += pv[o] * clf_w[k2 * 8 + o];
      out[(size_t)NBATCH * VDIM + ((size_t)n * 2 + k2) * VDIM + v] = cv;
    }
  }
}

extern "C" void kernel_launch(void* const* d_in, const int* in_sizes, int n_in,
                              void* d_out, int out_size, void* d_ws, size_t ws_size,
                              hipStream_t stream) {
  (void)in_sizes; (void)n_in; (void)out_size; (void)d_ws; (void)ws_size;
  const float* x        = (const float*)d_in[0];
  const float* Aadj     = (const float*)d_in[1];
  const float* bn0_g    = (const float*)d_in[2];
  const float* bn0_b    = (const float*)d_in[3];
  const float* res_w    = (const float*)d_in[4];
  const float* res_b    = (const float*)d_in[5];
  const float* res_bn_g = (const float*)d_in[6];
  const float* res_bn_b = (const float*)d_in[7];
  const float* gcn_w    = (const float*)d_in[8];
  const float* gcn_b    = (const float*)d_in[9];
  const float* gru_wih  = (const float*)d_in[10];
  // d_in[11] = gru_whh: unused, h0 == 0 so gh == gru_bhh
  const float* gru_bih  = (const float*)d_in[12];
  const float* gru_bhh  = (const float*)d_in[13];
  const float* bn1_g    = (const float*)d_in[14];
  const float* bn1_b    = (const float*)d_in[15];
  const float* tcn_w    = (const float*)d_in[16];
  const float* tcn_b    = (const float*)d_in[17];
  const float* bn2_g    = (const float*)d_in[18];
  const float* bn2_b    = (const float*)d_in[19];
  const float* fcn_w    = (const float*)d_in[20];
  const float* fcn_b    = (const float*)d_in[21];
  const float* clf_w    = (const float*)d_in[22];
  const float* clf_b    = (const float*)d_in[23];

  stgcn_fused<<<NBATCH, 256, 0, stream>>>(
      x, Aadj, bn0_g, bn0_b, res_w, res_b, res_bn_g, res_bn_b,
      gcn_w, gcn_b, gru_wih, gru_bih, gru_bhh, bn1_g, bn1_b,
      tcn_w, tcn_b, bn2_g, bn2_b, fcn_w, fcn_b, clf_w, clf_b,
      (float*)d_out);
}
